// GNN_10239202033930
// MI455X (gfx1250) — matrix operand order repaired and checked by reading, not compiled, not run on hardware
//
#include <hip/hip_runtime.h>
#include <hip/hip_bf16.h>

// ---------------- problem constants (match reference) ----------------
#define N_NODES  10000
#define N_EDGES  20000
#define G_GRAPHS 32
#define F_IN     42
#define FE_IN    6
#define H_DIM    64
#define L_LAYERS 3
#define NCHUNK   130   // 128 Wnn chunks (K=4096) + 2 bnn-fold chunks (K=64)

// ---------------- WMMA types (CDNA5 wave32, 16x16x32 bf16) ----------------
typedef __attribute__((ext_vector_type(16))) __bf16    v16bf;
typedef __attribute__((ext_vector_type(2)))  __bf16    v2bf;
typedef __attribute__((ext_vector_type(8)))  float     v8f;
typedef __attribute__((ext_vector_type(4)))  unsigned  u32x4;   // trivial 16B vector

union Frag16 {            // 16 bf16 = one WMMA A or B operand (8 VGPRs)
  unsigned u[8];
  u32x4    q[2];
  v2bf     h[8];
  v16bf    v;
};
union BFPair {
  unsigned       u;
  unsigned short s[2];
  v2bf           v;
};

__device__ __forceinline__ float bf2f(unsigned short s) {
  return __uint_as_float(((unsigned)s) << 16);
}
__device__ __forceinline__ unsigned short f2bf(float f) {
  unsigned b = __float_as_uint(f);
  b += 0x7fffu + ((b >> 16) & 1u);          // round-to-nearest-even-ish
  return (unsigned short)(b >> 16);
}

// ---------------- utility kernels ----------------
__global__ void zero_f32(float* __restrict__ p, int n) {
  int tid = blockIdx.x * blockDim.x + threadIdx.x;
  if (tid < n) p[tid] = 0.0f;
}

// h0 = x @ W_node + b_node  ->  bf16
__global__ void embed_nodes(const float* __restrict__ x, const float* __restrict__ W,
                            const float* __restrict__ b, unsigned short* __restrict__ h) {
  int tid = blockIdx.x * blockDim.x + threadIdx.x;
  if (tid >= N_NODES * H_DIM) return;
  int o = tid & 63, n = tid >> 6;
  const float* xr = x + n * F_IN;
  float s = b[o];
  for (int f = 0; f < F_IN; ++f) s += xr[f] * W[f * H_DIM + o];
  h[tid] = f2bf(s);
}

// e = edge_attr @ W_edge + b_edge  ->  bf16
__global__ void embed_edges(const float* __restrict__ ea, const float* __restrict__ W,
                            const float* __restrict__ b, unsigned short* __restrict__ e) {
  int tid = blockIdx.x * blockDim.x + threadIdx.x;
  if (tid >= N_EDGES * H_DIM) return;
  int o = tid & 63, n = tid >> 6;
  const float* er = ea + n * FE_IN;
  float s = b[o];
  for (int f = 0; f < FE_IN; ++f) s += er[f] * W[f * H_DIM + o];
  e[tid] = f2bf(s);
}

// Repack Wnn (+ bnn fold) into WMMA-B fragment order:
// Wb[((l*130 + c)*4 + ntile)*32 + lane][i] (16 contiguous bf16 per lane).
// B layout (32x16 bf16): lane<16 -> col n=lane, K={0..7,16..23}; lane>=16 -> col n=lane-16, K={8..15,24..31}.
// K index j = c*32 + kloc; for c<128: k=c>>1, h=(c&1)*32+kloc, elem = Wnn[l][k][h*64+n];
// for c>=128 (bias fold): h=(c-128)*32+kloc, elem = bnn[l][h*64+n].
__global__ void prep_wnn(const float* __restrict__ Wnn, const float* __restrict__ bnn,
                         unsigned short* __restrict__ Wb) {
  int tid = blockIdx.x * blockDim.x + threadIdx.x;
  if (tid >= L_LAYERS * NCHUNK * 2048) return;
  int i    = tid & 15;
  int lane = (tid >> 4) & 31;
  int tway = (tid >> 9) & 3;
  int lc   = tid >> 11;            // l*130 + c
  int c    = lc % NCHUNK;
  int l    = lc / NCHUNK;
  int n    = tway * 16 + (lane & 15);
  int kloc = (i >> 3) * 16 + (lane >> 4) * 8 + (i & 7);
  float val;
  if (c < 128) {
    int k = c >> 1;
    int h = (c & 1) * 32 + kloc;
    val = Wnn[(l * 64 + k) * 4096 + h * 64 + n];
  } else {
    int h = (c - 128) * 32 + kloc;
    val = bnn[l * 4096 + h * 64 + n];
  }
  Wb[tid] = f2bf(val);
}

// Repack root[l] (64x64) into B fragments: 2 K-chunks, 4 N-tiles per layer.
__global__ void prep_root(const float* __restrict__ root, unsigned short* __restrict__ rootB) {
  int tid = blockIdx.x * blockDim.x + threadIdx.x;
  if (tid >= L_LAYERS * 2 * 2048) return;
  int i    = tid & 15;
  int lane = (tid >> 4) & 31;
  int tway = (tid >> 9) & 3;
  int lc   = tid >> 11;            // l*2 + c
  int c    = lc & 1;
  int l    = lc >> 1;
  int n    = tway * 16 + (lane & 15);
  int kloc = (i >> 3) * 16 + (lane >> 4) * 8 + (i & 7);
  rootB[tid] = f2bf(root[l * 4096 + (c * 32 + kloc) * 64 + n]);
}

// ---------------- fused edge-net + message GEMM (the 10.5 GFLOP/layer part) ----------------
// Block = 16 edges, 4 waves = 4 N-tiles of 16. K = 4160 total, walked as k=0..63
// (two 32-chunks per k sharing the edge-feature scalar e[e,k]) + 2 bias-fold chunks.
// The h_src operand values depend only on hblk (0 or 32), so they are hoisted:
// 4 b128 LDS loads per lane cover ALL chunks; per-chunk A-gen is 8 v_pk_mul_bf16.
__global__ void edge_msg(const unsigned short* __restrict__ h_bf,
                         const unsigned short* __restrict__ e_bf,
                         const int* __restrict__ src,
                         const int* __restrict__ dst,
                         const unsigned short* __restrict__ Wb,   // layer's fragments
                         float* __restrict__ agg) {
  __shared__ u32x4 smem[256];                        // 16x64 bf16 h_src + 16x64 bf16 e rows
  unsigned short* hs = (unsigned short*)smem;
  unsigned short* es = (unsigned short*)(smem + 128);
  const int tile = blockIdx.x;
  const int t    = threadIdx.x;
  const int lane = t & 31;
  const int wave = t >> 5;                           // N-tile index
  {
    int r = t >> 3, part = t & 7;                    // 16 rows x 8 x 16B
    int e = tile * 16 + r;
    int s = src[e];
    ((u32x4*)hs)[r * 8 + part] = ((const u32x4*)h_bf)[s * 8 + part];
    ((u32x4*)es)[r * 8 + part] = ((const u32x4*)e_bf)[e * 8 + part];
  }
  __syncthreads();
  const int m  = lane & 15;                          // edge row within tile
  const int hi = lane >> 4;                          // A-layout half select
  // 16-bit A layout: lane<16 -> K = {0..7, 16..23}; lane>=16 -> K = {8..15, 24..31}
  Frag16 h0, h1;                                     // hblk=0 and hblk=32 source fragments
  h0.q[0] = *(const u32x4*)&hs[m * 64 +  0 + hi * 8];
  h0.q[1] = *(const u32x4*)&hs[m * 64 + 16 + hi * 8];
  h1.q[0] = *(const u32x4*)&hs[m * 64 + 32 + hi * 8];
  h1.q[1] = *(const u32x4*)&hs[m * 64 + 48 + hi * 8];
  const unsigned short* erow = &es[m * 64];
  const unsigned short* wp   = Wb + (wave * 32 + lane) * 16;  // fragment base; chunk stride 2048
  v8f acc = {};
#pragma unroll 2
  for (int k = 0; k < 64; ++k) {
    const u32x4* bp0 = (const u32x4*)(wp + k * 4096);         // chunk 2k   (hblk=0)
    const u32x4* bp1 = (const u32x4*)(wp + k * 4096 + 2048);  // chunk 2k+1 (hblk=32)
    Frag16 b0, b1;
    b0.q[0] = bp0[0]; b0.q[1] = bp0[1];
    b1.q[0] = bp1[0]; b1.q[1] = bp1[1];
    BFPair ev;
    unsigned short evs = erow[k];
    ev.u = ((unsigned)evs << 16) | (unsigned)evs;             // splat bf16 pair
    Frag16 a0, a1;
#pragma unroll
    for (int j = 0; j < 8; ++j) {                             // 16x v_pk_mul_bf16
      a0.h[j] = h0.h[j] * ev.v;
      a1.h[j] = h1.h[j] * ev.v;
    }
    acc = __builtin_amdgcn_wmma_f32_16x16x32_bf16(false, a0.v, false, b0.v,
                                                  (short)0, acc, false, false);
    acc = __builtin_amdgcn_wmma_f32_16x16x32_bf16(false, a1.v, false, b1.v,
                                                  (short)0, acc, false, false);
  }
  {
    // bias-fold chunks 128/129: A = h_src directly (u = h)
    const u32x4* bp0 = (const u32x4*)(wp + 128 * 2048);
    const u32x4* bp1 = (const u32x4*)(wp + 129 * 2048);
    Frag16 b0, b1;
    b0.q[0] = bp0[0]; b0.q[1] = bp0[1];
    b1.q[0] = bp1[0]; b1.q[1] = bp1[1];
    acc = __builtin_amdgcn_wmma_f32_16x16x32_bf16(false, h0.v, false, b0.v,
                                                  (short)0, acc, false, false);
    acc = __builtin_amdgcn_wmma_f32_16x16x32_bf16(false, h1.v, false, b1.v,
                                                  (short)0, acc, false, false);
  }
  // C layout: VGPR i, lane<16 -> (M=i, N=lane); lane>=16 -> (M=8+i, N=lane-16)
  const int n = wave * 16 + m;
#pragma unroll
  for (int i = 0; i < 8; ++i) {
    int mm = hi ? (8 + i) : i;
    int e  = tile * 16 + mm;
    unsafeAtomicAdd(&agg[dst[e] * 64 + n], acc[i]);
  }
}

// h_new = relu(h @ root[l] + agg + bias[l]) with WMMA (K=64 -> 2 chunks)
__global__ void node_update(const unsigned short* __restrict__ h_in,
                            const unsigned short* __restrict__ rootB, // layer's 2x4x32x16
                            const float* __restrict__ agg,
                            const float* __restrict__ bias,
                            unsigned short* __restrict__ h_out) {
  __shared__ u32x4 smem[128];                         // 16x64 bf16 node rows
  unsigned short* hs = (unsigned short*)smem;
  const int tile = blockIdx.x;
  const int t    = threadIdx.x;
  const int lane = t & 31;
  const int wave = t >> 5;
  {
    int r = t >> 3, part = t & 7;
    ((u32x4*)hs)[r * 8 + part] = ((const u32x4*)h_in)[(tile * 16 + r) * 8 + part];
  }
  __syncthreads();
  const int m  = lane & 15;
  const int hi = lane >> 4;
  v8f acc = {};
#pragma unroll
  for (int c = 0; c < 2; ++c) {
    const int s0 = c * 32 + hi * 8;
    const int s1 = c * 32 + 16 + hi * 8;
    Frag16 a, b;
    a.q[0] = *(const u32x4*)&hs[m * 64 + s0];
    a.q[1] = *(const u32x4*)&hs[m * 64 + s1];
    const u32x4* bp = (const u32x4*)&rootB[((c * 4 + wave) * 32 + lane) * 16];
    b.q[0] = bp[0]; b.q[1] = bp[1];
    acc = __builtin_amdgcn_wmma_f32_16x16x32_bf16(false, a.v, false, b.v,
                                                  (short)0, acc, false, false);
  }
  const int n = wave * 16 + m;
#pragma unroll
  for (int i = 0; i < 8; ++i) {
    int mm   = hi ? (8 + i) : i;
    int node = tile * 16 + mm;
    float v  = acc[i] + agg[node * 64 + n] + bias[n];
    v = v > 0.0f ? v : 0.0f;
    h_out[node * 64 + n] = f2bf(v);
  }
}

// global mean pool: scatter sums + counts
__global__ void pool_scatter(const unsigned short* __restrict__ h, const int* __restrict__ batch,
                             float* __restrict__ pooled, float* __restrict__ cnt) {
  int tid = blockIdx.x * blockDim.x + threadIdx.x;
  if (tid >= N_NODES * H_DIM) return;
  int o = tid & 63, n = tid >> 6;
  int g = batch[n];
  unsafeAtomicAdd(&pooled[g * H_DIM + o], bf2f(h[tid]));
  if (o == 0) unsafeAtomicAdd(&cnt[g], 1.0f);
}

// MLP head over 32 graphs: relu(pooled@W1+b1)@W2+b2
__global__ void head(const float* __restrict__ pooled, const float* __restrict__ cnt,
                     const float* __restrict__ W1, const float* __restrict__ b1,
                     const float* __restrict__ W2, const float* __restrict__ b2,
                     float* __restrict__ out) {
  int g = threadIdx.x;
  if (g >= G_GRAPHS) return;
  float c = cnt[g]; c = c > 1.0f ? c : 1.0f;
  float pr[H_DIM];
  for (int h = 0; h < H_DIM; ++h) pr[h] = pooled[g * H_DIM + h] / c;
  float acc = b2[0];
  for (int j = 0; j < 32; ++j) {
    float s = b1[j];
    for (int h = 0; h < H_DIM; ++h) s += pr[h] * W1[h * 32 + j];
    s = s > 0.0f ? s : 0.0f;
    acc += s * W2[j];
  }
  out[g] = acc;
}

// ---------------- launcher ----------------
extern "C" void kernel_launch(void* const* d_in, const int* in_sizes, int n_in,
                              void* d_out, int out_size, void* d_ws, size_t ws_size,
                              hipStream_t stream) {
  const float* x         = (const float*)d_in[0];
  const float* edge_attr = (const float*)d_in[1];
  const int*   edge_idx  = (const int*)d_in[2];
  const int*   batch     = (const int*)d_in[3];
  const float* W_node    = (const float*)d_in[4];
  const float* b_node    = (const float*)d_in[5];
  const float* W_edge    = (const float*)d_in[6];
  const float* b_edge    = (const float*)d_in[7];
  const float* Wnn       = (const float*)d_in[8];
  const float* bnn       = (const float*)d_in[9];
  const float* root      = (const float*)d_in[10];
  const float* bias      = (const float*)d_in[11];
  const float* W1        = (const float*)d_in[12];
  const float* b1        = (const float*)d_in[13];
  const float* W2        = (const float*)d_in[14];
  const float* b2        = (const float*)d_in[15];
  const int* src = edge_idx;
  const int* dst = edge_idx + N_EDGES;
  float* out = (float*)d_out;

  // workspace carve (all sizes 256B-aligned)
  char* p = (char*)d_ws;
  unsigned short* hA     = (unsigned short*)p;  p += N_NODES * H_DIM * 2;          // 1,280,000
  unsigned short* hB     = (unsigned short*)p;  p += N_NODES * H_DIM * 2;          // 1,280,000
  unsigned short* eb     = (unsigned short*)p;  p += N_EDGES * H_DIM * 2;          // 2,560,000
  unsigned short* WbAll  = (unsigned short*)p;  p += L_LAYERS * NCHUNK * 2048 * 2; // 1,597,440
  unsigned short* rootB  = (unsigned short*)p;  p += L_LAYERS * 2 * 2048 * 2;      //    24,576
  float*          agg    = (float*)p;           p += N_NODES * H_DIM * 4;          // 2,560,000
  float*          pooled = (float*)p;           p += G_GRAPHS * H_DIM * 4;         //     8,192
  float*          cnt    = (float*)p;           p += G_GRAPHS * 4;                 //       128

  (void)in_sizes; (void)n_in; (void)out_size; (void)ws_size;

  prep_wnn  <<<(L_LAYERS * NCHUNK * 2048 + 255) / 256, 256, 0, stream>>>(Wnn, bnn, WbAll);
  prep_root <<<(L_LAYERS * 2 * 2048 + 255) / 256,      256, 0, stream>>>(root, rootB);
  embed_nodes<<<(N_NODES * H_DIM + 255) / 256,         256, 0, stream>>>(x, W_node, b_node, hA);
  embed_edges<<<(N_EDGES * H_DIM + 255) / 256,         256, 0, stream>>>(edge_attr, W_edge, b_edge, eb);
  zero_f32  <<<(G_GRAPHS * H_DIM + G_GRAPHS + 255) / 256, 256, 0, stream>>>(pooled,
                                                     G_GRAPHS * H_DIM + G_GRAPHS);

  unsigned short* hcur = hA;
  unsigned short* hnext = hB;
  for (int l = 0; l < L_LAYERS; ++l) {
    zero_f32   <<<(N_NODES * H_DIM + 255) / 256, 256, 0, stream>>>(agg, N_NODES * H_DIM);
    edge_msg   <<<N_EDGES / 16, 128, 0, stream>>>(hcur, eb, src, dst,
                                                  WbAll + l * NCHUNK * 2048, agg);
    node_update<<<N_NODES / 16, 128, 0, stream>>>(hcur, rootB + l * 2 * 2048, agg,
                                                  bias + l * H_DIM, hnext);
    unsigned short* tmp = hcur; hcur = hnext; hnext = tmp;
  }

  pool_scatter<<<(N_NODES * H_DIM + 255) / 256, 256, 0, stream>>>(hcur, batch, pooled, cnt);
  head<<<1, 32, 0, stream>>>(pooled, cnt, W1, b1, W2, b2, out);
}
